// OneScalePredictor_773094113450
// MI455X (gfx1250) — matrix operand order repaired and checked by reading, not compiled, not run on hardware
//
#include <hip/hip_runtime.h>
#include <hip/hip_bf16.h>

// ---------------------------------------------------------------------------
// OneScalePredictor for MI455X (gfx1250, wave32, WMMA, async-LDS staging).
//
// - Sparse-conv and up_l2 GEMMs on v_wmma_f32_16x16x32_bf16 (fp32 accumulate).
// - Weights pre-transposed/converted to bf16 once; per-tap staging is a
//   contiguous 8KB async copy (global_load_async_to_lds_b128, ASYNCcnt),
//   double-buffered in LDS -> copy of tap k+1 overlaps WMMAs of tap k.
// - Feature maps consumed only as WMMA operands are stored bf16 (halves the
//   27-way L2 gather traffic; identical numerics since they fed bf16 WMMA
//   anyway).  f32 kept where VALU math / residuals need it.
// - Neighbor-mask scaling done as bf16 vector mul (V_PK_MUL_BF16 path).
// ---------------------------------------------------------------------------

typedef __attribute__((ext_vector_type(16))) __bf16 v16bf;
typedef __attribute__((ext_vector_type(8)))  __bf16 v8bf;
typedef __attribute__((ext_vector_type(8)))  float  v8f;

#define C_     64
#define K_     27
#define LOG2E_ 1.4426950408889634f

// 16-byte async global->LDS copy (CDNA5; tracked by ASYNCcnt).
__device__ __forceinline__ void async_copy16(unsigned lds_addr, const void* gbase,
                                             unsigned goff) {
    asm volatile("global_load_async_to_lds_b128 %0, %1, %2"
                 :: "v"(lds_addr), "v"(goff), "s"((unsigned long long)(uintptr_t)gbase)
                 : "memory");
}
__device__ __forceinline__ void wait_async0() {
    asm volatile("s_wait_asynccnt 0x0" ::: "memory");
}

// ---------------------------------------------------------------------------
// Sparse conv: y[n] = prelu( sum_k (x[nbr[n,k]] * m[n,k]) @ w[k] + b (+res), a)
// x is bf16, w pre-transposed bf16 [tap][cout][cin].  Block = 8 waves; each
// wave owns a 16x64 tile (4 f32 C-tiles).  Per tap: 8 B-frags from LDS,
// 2 A-frags from bf16 gathers, 8 WMMAs.  LDS double-buffered, one barrier
// per tap; next tap's weights stream in asynchronously.
// ---------------------------------------------------------------------------
__global__ __launch_bounds__(256) void sconv_bf16wmma_kernel(
    const __bf16* __restrict__ xin,            // [N][64] bf16 gather source
    const __bf16* __restrict__ wT,             // [27][64 cout][64 cin] bf16
    const float* __restrict__ bias, const float* __restrict__ alpha_p,
    const float* __restrict__ residual,        // optional f32 [N][64]
    const int* __restrict__ nbr_idx, const float* __restrict__ nbr_mask,
    float* __restrict__ yf,                    // optional f32 out
    __bf16* __restrict__ ybf,                  // optional bf16 out
    int N)
{
    __shared__ __bf16 ldsW[2][C_ * C_];        // 2 x 8KB, w[k]^T per tap
    const int tid  = threadIdx.x;
    const int wv   = tid >> 5;
    const int lane = tid & 31;
    const int lc   = lane & 15;
    const int kh   = lane >> 4;
    const int rowbase = blockIdx.x * 128 + wv * 16;
    const int myrow   = rowbase + lc;
    const bool rv     = myrow < N;
    const float alpha = *alpha_p;

    const unsigned ldsBase0 = (unsigned)(uintptr_t)&ldsW[0][0];
    const unsigned ldsBase1 = (unsigned)(uintptr_t)&ldsW[1][0];

    // each thread streams 32B of the 8KB tap-weight block
    const unsigned soff = (unsigned)tid * 16u;
    auto stage = [&](int tap, int buf) {
        const void* g = wT + (size_t)tap * (C_ * C_);
        const unsigned l = (buf ? ldsBase1 : ldsBase0);
        async_copy16(l + soff,         g, soff);
        async_copy16(l + soff + 4096u, g, soff + 4096u);
    };

    stage(0, 0);

    v8f acc[4] = {};

    for (int tap = 0; tap < K_; ++tap) {
        const int cur = tap & 1;
        wait_async0();
        __syncthreads();                       // weights for `tap` visible
        if (tap + 1 < K_) stage(tap + 1, cur ^ 1);   // overlaps WMMAs below

        int   gi = 0;
        float m  = 0.0f;
        if (rv) {
            gi = nbr_idx[(size_t)myrow * K_ + tap];
            m  = nbr_mask[(size_t)myrow * K_ + tap];
        }
        const __bf16* xb = xin + (size_t)gi * C_;

        // preload all 8 B fragments (batches the ds waits before the WMMAs)
        const __bf16* lw = &ldsW[cur][0];
        v16bf bfr[8];
#pragma unroll
        for (int cc = 0; cc < 2; ++cc)
#pragma unroll
            for (int t = 0; t < 4; ++t) {
                const __bf16* wr = lw + (t * 16 + lc) * C_ + cc * 32 + kh * 8;
                v16bf f;
#pragma unroll
                for (int j = 0; j < 16; ++j) f[j] = wr[j < 8 ? j : j + 8];
                bfr[cc * 4 + t] = f;
            }

        const __bf16 mb = (__bf16)m;
#pragma unroll
        for (int cc = 0; cc < 2; ++cc) {
            const int kb = cc * 32;
            // bf16 vector scale -> V_PK_MUL_BF16 path
            const v8bf s0 = *(const v8bf*)(xb + kb + kh * 8)      * mb;
            const v8bf s1 = *(const v8bf*)(xb + kb + 16 + kh * 8) * mb;
            v16bf a;
#pragma unroll
            for (int j = 0; j < 8; ++j) a[j]     = s0[j];
#pragma unroll
            for (int j = 0; j < 8; ++j) a[8 + j] = s1[j];
#pragma unroll
            for (int t = 0; t < 4; ++t)
                acc[t] = __builtin_amdgcn_wmma_f32_16x16x32_bf16(
                    false, a, false, bfr[cc * 4 + t], (short)0, acc[t], false, false);
        }
    }

    // epilogue: bias (+residual) + PReLU; optional f32 / bf16 stores
#pragma unroll
    for (int t = 0; t < 4; ++t) {
        const int col = t * 16 + lc;
        const float bv = bias[col];
#pragma unroll
        for (int r = 0; r < 8; ++r) {
            const int row = rowbase + r + kh * 8;
            if (row < N) {
                float v = acc[t][r] + bv;
                if (residual) v += residual[(size_t)row * C_ + col];
                v = v >= 0.0f ? v : alpha * v;
                if (yf)  yf [(size_t)row * C_ + col] = v;
                if (ybf) ybf[(size_t)row * C_ + col] = (__bf16)v;
            }
        }
    }
}

// ---------------------------------------------------------------------------
// up = prelu([rec | cur_bin] @ up_l1w + b).  [N,72]x[72,64] VALU (3.7 GFLOP).
// Writes f32 (residual consumer) + bf16 mirror (gather consumer).
// ---------------------------------------------------------------------------
__global__ __launch_bounds__(256) void upl1_kernel(
    const float* __restrict__ rec, const float* __restrict__ cur_bin,
    const float* __restrict__ wT,  // pre-transposed [64 co][72 ci]
    const float* __restrict__ b, const float* __restrict__ alpha_p,
    float* __restrict__ u, __bf16* __restrict__ ubf, int N)
{
    __shared__ float lw[64 * 72];
    const int tid = threadIdx.x;
    for (int i = tid; i < 64 * 72; i += 256) lw[i] = wT[i];   // contiguous
    __syncthreads();
    const int n = blockIdx.x * 256 + tid;
    if (n >= N) return;

    float in[72];
#pragma unroll
    for (int i = 0; i < 64; ++i) in[i] = rec[(size_t)n * 64 + i];
#pragma unroll
    for (int j = 0; j < 8; ++j) in[64 + j] = cur_bin[(size_t)n * 8 + j];
    const float alpha = *alpha_p;

    for (int co = 0; co < 64; ++co) {
        float acc = b[co];
#pragma unroll
        for (int ci = 0; ci < 72; ++ci) acc += in[ci] * lw[co * 72 + ci];
        acc = acc >= 0.0f ? acc : alpha * acc;
        u  [(size_t)n * 64 + co] = acc;
        ubf[(size_t)n * 64 + co] = (__bf16)acc;
    }
}

// ---------------------------------------------------------------------------
// out[n,8,64] = (u @ up_l2w + b).reshape(N,8,64) * cur_bin[:,:,None]
// A from bf16 mirror; B (pre-transposed bf16 [512 co][64 ci]) register-
// resident per wave; mask group per wave == wv; fused masked store.
// ---------------------------------------------------------------------------
__global__ __launch_bounds__(256) void upl2_wmma_kernel(
    const __bf16* __restrict__ u, const __bf16* __restrict__ wT,  // [512][64]
    const float* __restrict__ bias, const float* __restrict__ cur_bin,
    float* __restrict__ out, int N)
{
    const int tid  = threadIdx.x;
    const int wv   = tid >> 5;
    const int lane = tid & 31;
    const int lc   = lane & 15;
    const int kh   = lane >> 4;
    const int rowbase = blockIdx.x * 16;

    v16bf bfrag[8];
#pragma unroll
    for (int cc = 0; cc < 2; ++cc)
#pragma unroll
        for (int t = 0; t < 4; ++t) {
            const int c = wv * 64 + t * 16 + lc;
            const __bf16* wp = wT + (size_t)c * 64 + cc * 32 + kh * 8;
            v16bf f;
#pragma unroll
            for (int j = 0; j < 16; ++j) f[j] = wp[j < 8 ? j : j + 8];
            bfrag[cc * 4 + t] = f;
        }

    const int arow = rowbase + lc;
    const __bf16* xb = u + (size_t)(arow < N ? arow : 0) * C_;

    v8f acc[4] = {};
#pragma unroll
    for (int cc = 0; cc < 2; ++cc) {
        const int kb = cc * 32;
        const v8bf g0 = *(const v8bf*)(xb + kb + kh * 8);
        const v8bf g1 = *(const v8bf*)(xb + kb + 16 + kh * 8);
        v16bf a;
#pragma unroll
        for (int j = 0; j < 8; ++j) a[j]     = g0[j];
#pragma unroll
        for (int j = 0; j < 8; ++j) a[8 + j] = g1[j];
#pragma unroll
        for (int t = 0; t < 4; ++t)
            acc[t] = __builtin_amdgcn_wmma_f32_16x16x32_bf16(
                false, a, false, bfrag[cc * 4 + t], (short)0, acc[t], false, false);
    }

#pragma unroll
    for (int t = 0; t < 4; ++t) {
        const int col = wv * 64 + t * 16 + lc;
        const float bv = bias[col];
#pragma unroll
        for (int r = 0; r < 8; ++r) {
            const int row = rowbase + r + kh * 8;
            if (row < N) {
                const float mk = cur_bin[(size_t)row * 8 + wv];  // col>>6 == wv
                out[(size_t)row * 512 + col] = (acc[t][r] + bv) * mk;
            }
        }
    }
}

// ---------------------------------------------------------------------------
// Fused pred-linear (64->255) + log-softmax CE + per-batch normalization.
// lw pre-transposed [255 c][64 ci] -> contiguous LDS staging per 64-col chunk.
// ---------------------------------------------------------------------------
__global__ __launch_bounds__(256) void pred_loss_kernel(
    const float* __restrict__ feat, const float* __restrict__ lwT, // [255][64]
    const float* __restrict__ lb, const float* __restrict__ cur_bin,
    const int* __restrict__ batch_idx, const int* __restrict__ points_num,
    float* __restrict__ loss_out, int N, int Bn)
{
    __shared__ float lw[64 * 64];
    const int tid = threadIdx.x;
    const int n = blockIdx.x * 256 + tid;
    const bool valid = n < N;

    float xr[64];
#pragma unroll
    for (int i = 0; i < 64; ++i) xr[i] = 0.0f;
    if (valid) {
#pragma unroll
        for (int i = 0; i < 16; ++i) {
            const float4 v = ((const float4*)(feat + (size_t)n * 64))[i];
            xr[4 * i + 0] = v.x; xr[4 * i + 1] = v.y;
            xr[4 * i + 2] = v.z; xr[4 * i + 3] = v.w;
        }
    }
    int label = 0;
    if (valid) {
        int code = 0;
#pragma unroll
        for (int j = 0; j < 8; ++j) code += ((int)cur_bin[(size_t)n * 8 + j]) << j;
        label = code - 1;
    }

    float m = -3.402823466e38f, s = 0.0f, zlab = 0.0f;
    for (int c0 = 0; c0 < 255; c0 += 64) {
        const int nc = (255 - c0) < 64 ? (255 - c0) : 64;
        __syncthreads();
        for (int i = tid; i < nc * 64; i += 256) lw[i] = lwT[(size_t)c0 * 64 + i];
        __syncthreads();
        for (int cl = 0; cl < nc; ++cl) {
            float z = lb[c0 + cl];
#pragma unroll
            for (int ci = 0; ci < 64; ++ci) z += xr[ci] * lw[cl * 64 + ci];
            if (c0 + cl == label) zlab = z;
            const float mn = fmaxf(m, z);
            s = s * __expf(m - mn) + __expf(z - mn);
            m = mn;
        }
    }

    float contrib = 0.0f;
    if (valid) {
        const float lse = m + __logf(s);
        const float ce  = -(zlab - lse);
        const float pn  = (float)points_num[batch_idx[n]];
        contrib = (ce / pn) * (LOG2E_ / (float)Bn);
    }
#pragma unroll
    for (int off = 16; off > 0; off >>= 1)
        contrib += __shfl_down(contrib, off, 32);
    if ((tid & 31) == 0) atomicAdd(loss_out, contrib);
}

// --------------------------- prep kernels (tiny) ---------------------------
__global__ void f32_to_bf16_kernel(const float* __restrict__ in,
                                   __bf16* __restrict__ out, int n) {
    const int i = blockIdx.x * 256 + threadIdx.x;
    if (i < n) out[i] = (__bf16)in[i];
}
// out[t][c][r] = (bf16) in[t][r][c]   (compile-time dims -> no integer div)
template <int TT, int RR, int CC>
__global__ void transpose_to_bf16_kernel(const float* __restrict__ in,
                                         __bf16* __restrict__ out) {
    const int i = blockIdx.x * 256 + threadIdx.x;
    if (i >= TT * RR * CC) return;
    const int t = i / (RR * CC);
    const int rem = i - t * (RR * CC);
    const int r = rem / CC;
    const int c = rem - r * CC;
    out[(size_t)t * RR * CC + (size_t)c * RR + r] = (__bf16)in[i];
}
// out[c][r] = in[r][c]
template <int RR, int CC>
__global__ void transpose_f32_kernel(const float* __restrict__ in,
                                     float* __restrict__ out) {
    const int i = blockIdx.x * 256 + threadIdx.x;
    if (i >= RR * CC) return;
    const int r = i / CC, c = i - r * CC;
    out[(size_t)c * RR + r] = in[i];
}
__global__ void zero_loss_kernel(float* p) { *p = 0.0f; }

// ---------------------------------------------------------------------------
extern "C" void kernel_launch(void* const* d_in, const int* in_sizes, int n_in,
                              void* d_out, int out_size, void* d_ws, size_t ws_size,
                              hipStream_t stream) {
    const float* x        = (const float*)d_in[0];
    const float* cur_bin  = (const float*)d_in[2];
    const float* dec_w1   = (const float*)d_in[3];
    const float* dec_b1   = (const float*)d_in[4];
    const float* dec_a1   = (const float*)d_in[5];
    const float* dec_w2   = (const float*)d_in[6];
    const float* dec_b2   = (const float*)d_in[7];
    const float* dec_a2   = (const float*)d_in[8];
    const float* pred_w   = (const float*)d_in[9];
    const float* pred_b   = (const float*)d_in[10];
    const float* pred_a   = (const float*)d_in[11];
    const float* pred_lw  = (const float*)d_in[12];
    const float* pred_lb  = (const float*)d_in[13];
    const float* up_l1w   = (const float*)d_in[14];
    const float* up_l1b   = (const float*)d_in[15];
    const float* up_a     = (const float*)d_in[16];
    const float* upb_w1   = (const float*)d_in[17];
    const float* upb_b1   = (const float*)d_in[18];
    const float* upb_a1   = (const float*)d_in[19];
    const float* upb_w2   = (const float*)d_in[20];
    const float* upb_b2   = (const float*)d_in[21];
    const float* upb_a2   = (const float*)d_in[22];
    const float* up_l2w   = (const float*)d_in[23];
    const float* up_l2b   = (const float*)d_in[24];
    const int*   nbr_idx  = (const int*)d_in[25];
    const float* nbr_mask = (const float*)d_in[26];
    const int*   batch_idx  = (const int*)d_in[27];
    const int*   points_num = (const int*)d_in[28];

    const int N  = in_sizes[0] / C_;
    const int Bn = in_sizes[28];
    const int NC = N * C_;
    const int WSZ = K_ * C_ * C_;                 // 27*4096 elements per conv

    float* out  = (float*)d_out;                  // [N,8,64] + loss scalar
    float* loss = out + (size_t)N * 512;

    // workspace layout
    char* p = (char*)d_ws;
    float*  fP   = (float*)p;  p += (size_t)NC * 4;   // rec
    float*  fQ   = (float*)p;  p += (size_t)NC * 4;   // t, then u
    __bf16* M1   = (__bf16*)p; p += (size_t)NC * 2;   // bf16 mirror ping
    __bf16* M2   = (__bf16*)p; p += (size_t)NC * 2;   // bf16 mirror pong
    __bf16* wTc[5];
    for (int i = 0; i < 5; ++i) { wTc[i] = (__bf16*)p; p += (size_t)WSZ * 2; }
    __bf16* wl2T = (__bf16*)p; p += (size_t)512 * 64 * 2;
    float*  wl1T = (float*)p;  p += (size_t)64 * 72 * 4;
    float*  lwT  = (float*)p;  p += (size_t)255 * 64 * 4;

    const dim3 blk(256);
    const dim3 gConv((N + 127) / 128);
    const dim3 gRow((N + 255) / 256);
    const dim3 gL2((N + 15) / 16);
    const dim3 gNC((NC + 255) / 256);
    const dim3 gW((WSZ + 255) / 256);

    // ---- prep (tiny, deterministic) ----
    zero_loss_kernel<<<dim3(1), dim3(1), 0, stream>>>(loss);
    f32_to_bf16_kernel<<<gNC, blk, 0, stream>>>(x, M1, NC);
    const float* wsrc[5] = {dec_w1, dec_w2, pred_w, upb_w1, upb_w2};
    for (int i = 0; i < 5; ++i)
        transpose_to_bf16_kernel<K_, C_, C_><<<gW, blk, 0, stream>>>(wsrc[i], wTc[i]);
    transpose_to_bf16_kernel<1, 64, 512><<<dim3((512 * 64 + 255) / 256), blk, 0, stream>>>(
        up_l2w, wl2T);
    transpose_f32_kernel<72, 64><<<dim3((64 * 72 + 255) / 256), blk, 0, stream>>>(
        up_l1w, wl1T);
    transpose_f32_kernel<64, 255><<<dim3((255 * 64 + 255) / 256), blk, 0, stream>>>(
        pred_lw, lwT);

    // ---- dec res_block ----
    sconv_bf16wmma_kernel<<<gConv, blk, 0, stream>>>(
        M1, wTc[0], dec_b1, dec_a1, nullptr, nbr_idx, nbr_mask,
        nullptr, M2, N);                                            // h (bf16)
    sconv_bf16wmma_kernel<<<gConv, blk, 0, stream>>>(
        M2, wTc[1], dec_b2, dec_a2, x, nbr_idx, nbr_mask,
        fP, M1, N);                                                 // rec

    // ---- pred branch ----
    sconv_bf16wmma_kernel<<<gConv, blk, 0, stream>>>(
        M1, wTc[2], pred_b, pred_a, nullptr, nbr_idx, nbr_mask,
        fQ, nullptr, N);                                            // t (f32)
    pred_loss_kernel<<<gRow, blk, 0, stream>>>(
        fQ, lwT, pred_lb, cur_bin, batch_idx, points_num, loss, N, Bn);

    // ---- up branch ----
    upl1_kernel<<<gRow, blk, 0, stream>>>(
        fP, cur_bin, wl1T, up_l1b, up_a, fQ, M2, N);                // u
    sconv_bf16wmma_kernel<<<gConv, blk, 0, stream>>>(
        M2, wTc[3], upb_b1, upb_a1, nullptr, nbr_idx, nbr_mask,
        nullptr, M1, N);                                            // h2 (bf16)
    sconv_bf16wmma_kernel<<<gConv, blk, 0, stream>>>(
        M1, wTc[4], upb_b2, upb_a2, fQ, nbr_idx, nbr_mask,
        nullptr, M2, N);                                            // u2 (bf16)
    upl2_wmma_kernel<<<gL2, blk, 0, stream>>>(
        M2, wl2T, up_l2b, cur_bin, out, N);
}
